// CoAttention_LBSA_10960756539886
// MI455X (gfx1250) — matrix operand order repaired and checked by reading, not compiled, not run on hardware
//
#include <hip/hip_runtime.h>
#include <hip/hip_bf16.h>
#include <math.h>

#define B_DIM 64
#define T_DIM 512
#define D_DIM 1024

typedef __attribute__((ext_vector_type(16))) __bf16 v16bf;
typedef __attribute__((ext_vector_type(8)))  float  v8f;
typedef __attribute__((ext_vector_type(4)))  int    v4i;

union FragU { v16bf v; uint4 q[2]; };

__device__ __forceinline__ unsigned short f32_to_bf16_rne(float f) {
  unsigned int u = __float_as_uint(f);
  u += 0x7FFFu + ((u >> 16) & 1u);   // round-to-nearest-even
  return (unsigned short)(u >> 16);
}

// A fragment (16x32 bf16, MxK): lane holds row M=lane%16; half 0 lanes carry
// K = k0+[0..7] and k0+[16..23], half 1 lanes carry K = k0+[8..15] and k0+[24..31].
__device__ __forceinline__ v16bf load_frag_a(const unsigned short* __restrict__ row,
                                             int k0, int half) {
  FragU r;
  r.q[0] = *(const uint4*)(row + k0 + (half ? 8 : 0));
  r.q[1] = *(const uint4*)(row + k0 + (half ? 24 : 16));
  return r.v;
}

// B fragment (32x16 bf16, KxN): lane holds column N=lane%16; half 0 lanes carry
// K = k0+[0..7] and k0+[16..23], half 1 lanes carry K = k0+[8..15] and k0+[24..31]. `row` points at the
// K-major source row for this lane's column (W^T row, or x row for GEMM2).
__device__ __forceinline__ v16bf load_frag_b(const unsigned short* __restrict__ row,
                                             int k0, int half) {
  FragU r;
  r.q[0] = *(const uint4*)(row + k0 + (half ? 8 : 0));
  r.q[1] = *(const uint4*)(row + k0 + (half ? 24 : 16));
  return r.v;
}

// D-matrix 16x16 f32 layout: VGPR r -> row M = r + 8*(lane/16), col N = lane%16.
__device__ __forceinline__ void store_tile_bf16(unsigned short* __restrict__ dst,
                                                int m0, int n0, int half, int lc,
                                                v8f acc) {
#pragma unroll
  for (int r = 0; r < 8; ++r)
    dst[(size_t)(m0 + r + 8 * half) * D_DIM + n0 + lc] = f32_to_bf16_rne(acc[r]);
}

// ---- 1) cast x (f32) -> bf16, row-major [B*T, D] ----
__global__ void k_convert_x(const float* __restrict__ x,
                            unsigned short* __restrict__ xb, int n) {
  int i = blockIdx.x * blockDim.x + threadIdx.x;
  if (i < n) xb[i] = f32_to_bf16_rne(x[i]);
}

// ---- 2) cast + transpose W (f32 [D,D]) -> bf16 W^T [D,D] ----
__global__ void k_convert_wt(const float* __restrict__ w,
                             unsigned short* __restrict__ wtb) {
  int i = blockIdx.x * blockDim.x + threadIdx.x;  // i = d*D + e
  int d = i / D_DIM, e = i - d * D_DIM;
  wtb[(size_t)e * D_DIM + d] = f32_to_bf16_rne(w[i]);
}

// ---- 3) GEMM1: XW[m,e] = sum_d xb[m,d] * W[d,e], output bf16 ----
// 32x32 register tile per wave (2 A-frags x 2 B-frags -> 4 WMMAs / K-step).
// Workgroup = 8 waves arranged 2(M) x 4(N) covering 64x128.
// grid = (B*T/64, D/128), block = 256.
__global__ void k_gemm1(const unsigned short* __restrict__ xb,
                        const unsigned short* __restrict__ wtb,
                        unsigned short* __restrict__ xwb) {
  const int wave = threadIdx.x >> 5;
  const int lane = threadIdx.x & 31;
  const int half = lane >> 4;
  const int lc   = lane & 15;
  const int m0 = blockIdx.x * 64  + (wave >> 2) * 32;
  const int n0 = blockIdx.y * 128 + (wave & 3) * 32;

  const unsigned short* rowA0 = xb  + (size_t)(m0 + lc) * D_DIM;
  const unsigned short* rowA1 = rowA0 + (size_t)16 * D_DIM;
  const unsigned short* rowB0 = wtb + (size_t)(n0 + lc) * D_DIM;
  const unsigned short* rowB1 = rowB0 + (size_t)16 * D_DIM;

  v8f acc00 = {0.f,0.f,0.f,0.f,0.f,0.f,0.f,0.f};
  v8f acc01 = acc00, acc10 = acc00, acc11 = acc00;

#pragma unroll 2
  for (int k0 = 0; k0 < D_DIM; k0 += 32) {
    // locality 3 -> WGP-scope prefetch (warms the near cache, not just GL2)
    __builtin_prefetch(rowA0 + k0 + 256, 0, 3);
    __builtin_prefetch(rowB0 + k0 + 256, 0, 3);
    v16bf a0 = load_frag_a(rowA0, k0, half);
    v16bf a1 = load_frag_a(rowA1, k0, half);
    v16bf b0 = load_frag_b(rowB0, k0, half);
    v16bf b1 = load_frag_b(rowB1, k0, half);
    acc00 = __builtin_amdgcn_wmma_f32_16x16x32_bf16(false, a0, false, b0, (short)0, acc00, false, false);
    acc01 = __builtin_amdgcn_wmma_f32_16x16x32_bf16(false, a0, false, b1, (short)0, acc01, false, false);
    acc10 = __builtin_amdgcn_wmma_f32_16x16x32_bf16(false, a1, false, b0, (short)0, acc10, false, false);
    acc11 = __builtin_amdgcn_wmma_f32_16x16x32_bf16(false, a1, false, b1, (short)0, acc11, false, false);
  }

  store_tile_bf16(xwb, m0,      n0,      half, lc, acc00);
  store_tile_bf16(xwb, m0,      n0 + 16, half, lc, acc01);
  store_tile_bf16(xwb, m0 + 16, n0,      half, lc, acc10);
  store_tile_bf16(xwb, m0 + 16, n0 + 16, half, lc, acc11);
}

// ---- 4) fused GEMM2 + bias + tanh + context reduction -> scores[B,T] ----
// eij[t,s] = sum_e XW[t,e]*x[s,e]; scores[t] = sum_s tanh(eij+bias[s])*cv[s]
// grid = (B, T/16), block = 256 (8 waves). Wave w owns s-tiles 4w..4w+3 and
// processes them concurrently so one A fragment feeds 4 WMMAs per K-step.
__global__ void k_scores(const unsigned short* __restrict__ xwb,
                         const unsigned short* __restrict__ xb,
                         const float* __restrict__ bias,
                         const float* __restrict__ cv,
                         float* __restrict__ scores) {
  __shared__ float sc[16];
  const int wave = threadIdx.x >> 5;
  const int lane = threadIdx.x & 31;
  const int half = lane >> 4;
  const int lc   = lane & 15;
  const int b  = blockIdx.x;
  const int t0 = blockIdx.y * 16;

  if (threadIdx.x < 16) sc[threadIdx.x] = 0.f;
  __syncthreads();

  const unsigned short* rowA = xwb + (size_t)(b * T_DIM + t0 + lc) * D_DIM;
  const unsigned short* rowB[4];
#pragma unroll
  for (int j = 0; j < 4; ++j)
    rowB[j] = xb + (size_t)(b * T_DIM + (wave * 4 + j) * 16 + lc) * D_DIM;

  v8f acc[4];
#pragma unroll
  for (int j = 0; j < 4; ++j) acc[j] = (v8f){0.f,0.f,0.f,0.f,0.f,0.f,0.f,0.f};

#pragma unroll 2
  for (int k0 = 0; k0 < D_DIM; k0 += 32) {
    __builtin_prefetch(rowA + k0 + 256, 0, 3);
    v16bf a = load_frag_a(rowA, k0, half);
#pragma unroll
    for (int j = 0; j < 4; ++j) {
      v16bf bb = load_frag_b(rowB[j], k0, half);
      acc[j] = __builtin_amdgcn_wmma_f32_16x16x32_bf16(false, a, false, bb,
                                                       (short)0, acc[j], false, false);
    }
  }

  float rowsum[8];
#pragma unroll
  for (int r = 0; r < 8; ++r) rowsum[r] = 0.f;
#pragma unroll
  for (int j = 0; j < 4; ++j) {
    const int s = (wave * 4 + j) * 16 + lc;   // this lane's column (N = s)
    const float bs  = bias[s];
    const float cvs = cv[s];
#pragma unroll
    for (int r = 0; r < 8; ++r)
      rowsum[r] += tanhf(acc[j][r] + bs) * cvs;
  }

  // reduce each row over the 16 lanes of its half-wave, then across waves in LDS
#pragma unroll
  for (int r = 0; r < 8; ++r) {
    float v = rowsum[r];
    v += __shfl_xor(v, 1, 32);
    v += __shfl_xor(v, 2, 32);
    v += __shfl_xor(v, 4, 32);
    v += __shfl_xor(v, 8, 32);
    if (lc == 0) atomicAdd(&sc[r + 8 * half], v);
  }
  __syncthreads();
  if (threadIdx.x < 16)
    scores[(size_t)b * T_DIM + t0 + threadIdx.x] = sc[threadIdx.x];
}

// ---- 5) masked softmax over T (one block of 512 threads per batch) ----
__global__ void k_softmax(const float* __restrict__ scores,
                          const unsigned char* __restrict__ mask,
                          float* __restrict__ a_out) {
  __shared__ float redmax[16];
  __shared__ float redsum[16];
  const int b = blockIdx.x;
  const int t = threadIdx.x;
  const int wave = t >> 5, lane = t & 31;

  const bool m = mask[(size_t)b * T_DIM + t] != 0;
  const float s = m ? scores[(size_t)b * T_DIM + t] : -INFINITY;

  float v = s;
  for (int off = 1; off < 32; off <<= 1) v = fmaxf(v, __shfl_xor(v, off, 32));
  if (lane == 0) redmax[wave] = v;
  __syncthreads();
  if (t < 16) {
    float w = redmax[t];
    for (int off = 1; off < 16; off <<= 1) w = fmaxf(w, __shfl_xor(w, off, 32));
    redmax[t] = w;
  }
  __syncthreads();
  const float mx = redmax[0];

  const float e = m ? __expf(s - mx) : 0.f;
  v = e;
  for (int off = 1; off < 32; off <<= 1) v += __shfl_xor(v, off, 32);
  if (lane == 0) redsum[wave] = v;
  __syncthreads();
  if (t < 16) {
    float w = redsum[t];
    for (int off = 1; off < 16; off <<= 1) w += __shfl_xor(w, off, 32);
    redsum[t] = w;
  }
  __syncthreads();
  a_out[(size_t)b * T_DIM + t] = e / redsum[0];
}

// ---- 6) out[b,d] = sum_t x[b,t,d] * a[b,t]  (f32, bandwidth-bound) ----
// Stages a[b,:] into LDS via the gfx1250 async-to-LDS path when available
// (GLOBAL_LOAD_ASYNC_TO_LDS_B128, tracked by ASYNCcnt), else plain LDS stores.
__global__ void k_outsum(const float* __restrict__ x,
                         const float* __restrict__ a,
                         float* __restrict__ out) {
  __shared__ float as[T_DIM];
  const int b = blockIdx.x;
  const int d = blockIdx.y * 256 + threadIdx.x;

#if defined(__gfx1250__) && __has_builtin(__builtin_amdgcn_global_load_async_to_lds_b128)
  // builtin signature (from toolchain): (v4i addrspace(1)*, v4i addrspace(3)*, imm offset, imm cpol)
  typedef __attribute__((address_space(1))) v4i gv4i;
  typedef __attribute__((address_space(3))) v4i lv4i;
  if (threadIdx.x < T_DIM / 4) {   // 128 lanes x 16B = 512 floats
    __builtin_amdgcn_global_load_async_to_lds_b128(
        (gv4i*)(a + (size_t)b * T_DIM + threadIdx.x * 4),
        (lv4i*)(as + threadIdx.x * 4),
        0, 0);
  }
  asm volatile("s_wait_asynccnt 0" ::: "memory");
  __syncthreads();
#else
  for (int t = threadIdx.x; t < T_DIM; t += 256) as[t] = a[(size_t)b * T_DIM + t];
  __syncthreads();
#endif

  const float* xp = x + (size_t)b * T_DIM * D_DIM + d;
  float acc = 0.f;
#pragma unroll 4
  for (int t = 0; t < T_DIM; ++t) acc += xp[(size_t)t * D_DIM] * as[t];
  out[(size_t)b * D_DIM + d] = acc;
}

extern "C" void kernel_launch(void* const* d_in, const int* in_sizes, int n_in,
                              void* d_out, int out_size, void* d_ws, size_t ws_size,
                              hipStream_t stream) {
  const float*         x    = (const float*)d_in[0];          // [B,T,D] f32
  const unsigned char* mask = (const unsigned char*)d_in[1];  // [B,T] bool
  const float*         w    = (const float*)d_in[2];          // [D,D] f32
  const float*         cv   = (const float*)d_in[3];          // [T,1] f32
  const float*         bias = (const float*)d_in[4];          // [T] f32

  float* out   = (float*)d_out;                         // [B,D]
  float* a_out = out + (size_t)B_DIM * D_DIM;           // [B,T]

  // workspace carve-up (~130 MB)
  char* ws = (char*)d_ws;
  unsigned short* xb  = (unsigned short*)ws; ws += (size_t)B_DIM * T_DIM * D_DIM * 2;
  unsigned short* xwb = (unsigned short*)ws; ws += (size_t)B_DIM * T_DIM * D_DIM * 2;
  unsigned short* wtb = (unsigned short*)ws; ws += (size_t)D_DIM * D_DIM * 2;
  float* scores = (float*)ws;

  const int nx = B_DIM * T_DIM * D_DIM;
  k_convert_x <<<nx / 256, 256, 0, stream>>>(x, xb, nx);
  k_convert_wt<<<(D_DIM * D_DIM) / 256, 256, 0, stream>>>(w, wtb);
  k_gemm1     <<<dim3(B_DIM * T_DIM / 64, D_DIM / 128), 256, 0, stream>>>(xb, wtb, xwb);
  k_scores    <<<dim3(B_DIM, T_DIM / 16), 256, 0, stream>>>(xwb, xb, bias, cv, scores);
  k_softmax   <<<B_DIM, T_DIM, 0, stream>>>(scores, mask, a_out);
  k_outsum    <<<dim3(B_DIM, D_DIM / 256), 256, 0, stream>>>(x, a_out, out);
}
